// CustomAttention_60679297958058
// MI455X (gfx1250) — matrix operand order repaired and checked by reading, not compiled, not run on hardware
//
#include <hip/hip_runtime.h>

#define NTOK   197
#define NPAD   224            // 14*16, 7*32
#define DMODEL 768
#define NHEAD  12
#define HDIM   64
#define BATCH  4
#define MROWS  788            // BATCH*NTOK
#define MPAD   800            // 50*16

typedef __attribute__((ext_vector_type(16))) _Float16 v16h;
typedef __attribute__((ext_vector_type(8)))  _Float16 v8h;
typedef __attribute__((ext_vector_type(8)))  float    v8f;

// ---------- workspace layout (bytes) ----------
static const size_t OFF_XH = 0;                                   // f16 [800][768]
static const size_t SZ_XH  = (size_t)MPAD * DMODEL * 2;
static const size_t OFF_QH = OFF_XH + SZ_XH;                      // f16 [48][224][64]
static const size_t SZ_HD1 = (size_t)BATCH * NHEAD * NPAD * HDIM * 2;
static const size_t OFF_KH = OFF_QH + SZ_HD1;                     // f16 [48][224][64]
static const size_t OFF_VT = OFF_KH + SZ_HD1;                     // f16 [48][64][224]  (V transposed)
static const size_t OFF_P  = OFF_VT + SZ_HD1;                     // f16 [48][224][224]
static const size_t SZ_P   = (size_t)BATCH * NHEAD * NPAD * NPAD * 2;
static const size_t ZERO_BYTES = OFF_P + SZ_P;
static const size_t OFF_WT = ZERO_BYTES;                          // f16 [3][768][768]  (W transposed: [dout][din])
static const size_t SZ_WT  = (size_t)3 * DMODEL * DMODEL * 2;
static const size_t OFF_S  = OFF_WT + SZ_WT;                      // f32 [48][224][224]

// ---------- WMMA fragment loaders (ISA 7.12.2 layouts, wave32) ----------
// All loads are two aligned 16-byte chunks per lane.

// A: 16x32 f16 tile, row-major source, leading dim ld (halves).
// lane<16: row=lane, K in {0..7, 16..23}; lane>=16: row=lane-16, K in {8..15, 24..31}
__device__ __forceinline__ v16h load_frag_a(const _Float16* base, int ld, int lane) {
  int row = lane & 15;
  int kb  = (lane >> 4) ? 8 : 0;
  const _Float16* p = base + (long)row * ld + kb;
  v8h lo = *(const v8h*)p;
  v8h hi = *(const v8h*)(p + 16);
  return __builtin_shufflevector(lo, hi, 0, 1, 2, 3, 4, 5, 6, 7,
                                 8, 9, 10, 11, 12, 13, 14, 15);
}

// B: logical 32x16 (K x N) where memory holds B^T row-major: elem(k,n) = base[n*ld + k].
// lane<16: col=lane, K in {0..7, 16..23}; lane>=16: col=lane-16, K in {8..15, 24..31}
__device__ __forceinline__ v16h load_frag_bt(const _Float16* base, int ld, int lane) {
  int col = lane & 15;
  int kb  = (lane >> 4) ? 8 : 0;
  const _Float16* p = base + (long)col * ld + kb;
  v8h lo = *(const v8h*)p;
  v8h hi = *(const v8h*)(p + 16);
  return __builtin_shufflevector(lo, hi, 0, 1, 2, 3, 4, 5, 6, 7,
                                 8, 9, 10, 11, 12, 13, 14, 15);
}

// ---------- kernels ----------
__global__ void k_zero(uint4* p, long n16) {
  long i = (long)blockIdx.x * blockDim.x + threadIdx.x;
  long stride = (long)gridDim.x * blockDim.x;
  uint4 z = {0u, 0u, 0u, 0u};
  for (; i < n16; i += stride) p[i] = z;
}

// X -> f16 (row-major), W{q,k,v} -> f16 transposed [dout][din].
__global__ void k_convert(const float* __restrict__ X,
                          const float* __restrict__ Wq,
                          const float* __restrict__ Wk,
                          const float* __restrict__ Wv,
                          _Float16* __restrict__ Xh,
                          _Float16* __restrict__ Wt) {
  const long NX = (long)MROWS * DMODEL;
  const long NW = (long)DMODEL * DMODEL;
  long i = (long)blockIdx.x * blockDim.x + threadIdx.x;
  long stride = (long)gridDim.x * blockDim.x;
  for (; i < NX; i += stride) {
    Xh[i] = (_Float16)X[i];
    if (i < NW) {
      long o  = i / DMODEL;          // dout
      long ii = i - o * DMODEL;      // din
      long src = ii * DMODEL + o;
      Wt[i]          = (_Float16)Wq[src];
      Wt[i + NW]     = (_Float16)Wk[src];
      Wt[i + 2 * NW] = (_Float16)Wv[src];
    }
  }
}

// One wave computes a 16x64 output tile (one head's width) of the QKV GEMM.
// A fragment reused across 4 accumulators per K-step.
__global__ void k_qkv(const _Float16* __restrict__ Xh,
                      const _Float16* __restrict__ Wt,
                      const float* __restrict__ bq,
                      const float* __restrict__ bk,
                      const float* __restrict__ bv,
                      _Float16* __restrict__ qh,
                      _Float16* __restrict__ kh,
                      _Float16* __restrict__ vt) {
  int lane  = threadIdx.x;
  int mt    = blockIdx.x;   // 0..49
  int h     = blockIdx.y;   // 0..11 (64-wide output chunk == head)
  int which = blockIdx.z;   // 0:Q 1:K 2:V
  const _Float16* W = Wt + (long)which * DMODEL * DMODEL;
  const float* bias = (which == 0) ? bq : (which == 1) ? bk : bv;

  v8f c0 = {}, c1 = {}, c2 = {}, c3 = {};
  const _Float16* arow = Xh + (long)mt * 16 * DMODEL;
  const _Float16* brow = W + (long)h * 64 * DMODEL;
  for (int kt = 0; kt < DMODEL / 32; ++kt) {
    v16h a = load_frag_a(arow + kt * 32, DMODEL, lane);
    if (kt + 1 < DMODEL / 32)
      __builtin_prefetch(arow + (kt + 1) * 32 + (lane & 15) * DMODEL, 0, 1);
    v16h b0 = load_frag_bt(brow + 0 * 16 * DMODEL + kt * 32, DMODEL, lane);
    v16h b1 = load_frag_bt(brow + 1 * 16 * DMODEL + kt * 32, DMODEL, lane);
    v16h b2 = load_frag_bt(brow + 2 * 16 * DMODEL + kt * 32, DMODEL, lane);
    v16h b3 = load_frag_bt(brow + 3 * 16 * DMODEL + kt * 32, DMODEL, lane);
    c0 = __builtin_amdgcn_wmma_f32_16x16x32_f16(false, a, false, b0, (short)0, c0, false, false);
    c1 = __builtin_amdgcn_wmma_f32_16x16x32_f16(false, a, false, b1, (short)0, c1, false, false);
    c2 = __builtin_amdgcn_wmma_f32_16x16x32_f16(false, a, false, b2, (short)0, c2, false, false);
    c3 = __builtin_amdgcn_wmma_f32_16x16x32_f16(false, a, false, b3, (short)0, c3, false, false);
  }

  int col  = lane & 15;
  int roff = (lane >> 4) ? 8 : 0;
#pragma unroll
  for (int s = 0; s < 4; ++s) {
    v8f c = (s == 0) ? c0 : (s == 1) ? c1 : (s == 2) ? c2 : c3;
    int dd   = s * 16 + col;
    float bv_ = bias[h * 64 + dd];
#pragma unroll
    for (int r = 0; r < 8; ++r) {
      int m = mt * 16 + r + roff;
      if (m < MROWS) {
        int bidx = m / NTOK;
        int n = m - bidx * NTOK;
        long bh = (long)(bidx * NHEAD + h);
        _Float16 val = (_Float16)(c[r] + bv_);
        if (which == 0)      qh[(bh * NPAD + n) * HDIM + dd] = val;
        else if (which == 1) kh[(bh * NPAD + n) * HDIM + dd] = val;
        else                 vt[(bh * HDIM + dd) * NPAD + n] = val;
      }
    }
  }
}

// S = Q K^T per head. One wave per 16-row block: A fragments (HD=64) loaded once,
// reused across all 14 column tiles -> 28 wmmas per wave.
__global__ void k_logits(const _Float16* __restrict__ qh,
                         const _Float16* __restrict__ kh,
                         float* __restrict__ S) {
  int lane = threadIdx.x;
  int it = blockIdx.x;   // 0..12
  int bh = blockIdx.y;   // 0..47
  const _Float16* qb = qh + (long)bh * NPAD * HDIM;
  const _Float16* kb = kh + (long)bh * NPAD * HDIM;
  float* Sb = S + (long)bh * NPAD * NPAD;

  v16h a0 = load_frag_a(qb + (long)it * 16 * HDIM + 0,  HDIM, lane);
  v16h a1 = load_frag_a(qb + (long)it * 16 * HDIM + 32, HDIM, lane);

  int col = lane & 15;
  int roff = (lane >> 4) ? 8 : 0;
#pragma unroll
  for (int jt = 0; jt < NPAD / 16; ++jt) {
    v16h b0 = load_frag_bt(kb + (long)jt * 16 * HDIM + 0,  HDIM, lane);
    v16h b1 = load_frag_bt(kb + (long)jt * 16 * HDIM + 32, HDIM, lane);
    v8f c = {};
    c = __builtin_amdgcn_wmma_f32_16x16x32_f16(false, a0, false, b0, (short)0, c, false, false);
    c = __builtin_amdgcn_wmma_f32_16x16x32_f16(false, a1, false, b1, (short)0, c, false, false);
#pragma unroll
    for (int r = 0; r < 8; ++r)
      Sb[(long)(it * 16 + r + roff) * NPAD + jt * 16 + col] = c[r];
  }
}

__device__ __forceinline__ float block_reduce_max(float v, float* red, int t) {
  red[t] = v; __syncthreads();
  for (int s = 128; s > 0; s >>= 1) {
    if (t < s) red[t] = fmaxf(red[t], red[t + s]);
    __syncthreads();
  }
  float r = red[0]; __syncthreads();
  return r;
}
__device__ __forceinline__ float block_reduce_sum(float v, float* red, int t) {
  red[t] = v; __syncthreads();
  for (int s = 128; s > 0; s >>= 1) {
    if (t < s) red[t] += red[t + s];
    __syncthreads();
  }
  float r = red[0]; __syncthreads();
  return r;
}

// Per (bh, row i): 3-branch softmax mix -> P (f16, padded cols zeroed).
// Branch 3: T[i,j] = Ws . tanh(k_i + q_j) + bs, softmax over j.
__global__ void k_combine(const float* __restrict__ S,
                          const _Float16* __restrict__ qh,
                          const _Float16* __restrict__ kh,
                          const float* __restrict__ Ws,
                          const float* __restrict__ bs,
                          _Float16* __restrict__ P) {
  int i  = blockIdx.x;   // 0..196
  int bh = blockIdx.y;   // 0..47
  int t  = threadIdx.x;  // 0..255
  __shared__ float red[256];
  __shared__ float ksh[HDIM];
  __shared__ float wsh[HDIM];
  const _Float16* qb = qh + (long)bh * NPAD * HDIM;
  const _Float16* kb = kh + (long)bh * NPAD * HDIM;
  if (t < HDIM) {
    ksh[t] = (float)kb[(long)i * HDIM + t];
    wsh[t] = Ws[t];
  }
  __syncthreads();

  bool valid = (t < NTOK);
  float s = valid ? S[(long)bh * NPAD * NPAD + (long)i * NPAD + t] : -1e30f;

  float t3 = -1e30f;
  if (valid) {
    const _Float16* qv = qb + (long)t * HDIM;
    float acc = 0.f;
#pragma unroll 8
    for (int d = 0; d < HDIM; ++d) acc += wsh[d] * tanhf(ksh[d] + (float)qv[d]);
    t3 = acc + bs[0];
  }

  float m1 = block_reduce_max(s, red, t);
  float m3 = block_reduce_max(t3, red, t);
  float m2 = m1 * 0.125f;   // max of scaled row = scaled max (positive scale)

  float e1 = valid ? __expf(s - m1) : 0.f;
  float e2 = valid ? __expf(s * 0.125f - m2) : 0.f;
  float e3 = valid ? __expf(t3 - m3) : 0.f;

  float z1 = block_reduce_sum(e1, red, t);
  float z2 = block_reduce_sum(e2, red, t);
  float z3 = block_reduce_sum(e3, red, t);

  if (t < NPAD) {
    float p = valid ? (e1 / z1 + e2 / z2 + e3 / z3) * (1.0f / 3.0f) : 0.f;
    P[(long)bh * NPAD * NPAD + (long)i * NPAD + t] = (_Float16)p;
  }
}

// ctx = P * V per head. One wave computes 16x64 (the whole head dim):
// 7 K-steps x 4 accumulators sharing each A fragment -> 28 wmmas per wave.
// V is stored transposed [d][j], so B fragments are contiguous per lane.
__global__ void k_ctx(const _Float16* __restrict__ P,
                      const _Float16* __restrict__ vt,
                      float* __restrict__ out) {
  int lane = threadIdx.x;
  int it = blockIdx.x;   // 0..12
  int bh = blockIdx.y;   // 0..47
  const _Float16* Pb = P + (long)bh * NPAD * NPAD + (long)it * 16 * NPAD;
  const _Float16* vb = vt + (long)bh * HDIM * NPAD;

  v8f c0 = {}, c1 = {}, c2 = {}, c3 = {};
#pragma unroll
  for (int kt = 0; kt < NPAD / 32; ++kt) {
    v16h a = load_frag_a(Pb + kt * 32, NPAD, lane);
    v16h b0 = load_frag_bt(vb + (long)(0 * 16) * NPAD + kt * 32, NPAD, lane);
    v16h b1 = load_frag_bt(vb + (long)(1 * 16) * NPAD + kt * 32, NPAD, lane);
    v16h b2 = load_frag_bt(vb + (long)(2 * 16) * NPAD + kt * 32, NPAD, lane);
    v16h b3 = load_frag_bt(vb + (long)(3 * 16) * NPAD + kt * 32, NPAD, lane);
    c0 = __builtin_amdgcn_wmma_f32_16x16x32_f16(false, a, false, b0, (short)0, c0, false, false);
    c1 = __builtin_amdgcn_wmma_f32_16x16x32_f16(false, a, false, b1, (short)0, c1, false, false);
    c2 = __builtin_amdgcn_wmma_f32_16x16x32_f16(false, a, false, b2, (short)0, c2, false, false);
    c3 = __builtin_amdgcn_wmma_f32_16x16x32_f16(false, a, false, b3, (short)0, c3, false, false);
  }

  int b = bh / NHEAD, h = bh - b * NHEAD;
  int col = lane & 15;
  int roff = (lane >> 4) ? 8 : 0;
#pragma unroll
  for (int s = 0; s < 4; ++s) {
    v8f c = (s == 0) ? c0 : (s == 1) ? c1 : (s == 2) ? c2 : c3;
#pragma unroll
    for (int r = 0; r < 8; ++r) {
      int n = it * 16 + r + roff;
      if (n < NTOK)
        out[((long)(b * NTOK + n)) * DMODEL + h * HDIM + s * 16 + col] = c[r];
    }
  }
}

extern "C" void kernel_launch(void* const* d_in, const int* in_sizes, int n_in,
                              void* d_out, int out_size, void* d_ws, size_t ws_size,
                              hipStream_t stream) {
  const float* X  = (const float*)d_in[0];
  const float* Wq = (const float*)d_in[1];
  const float* bq = (const float*)d_in[2];
  const float* Wk = (const float*)d_in[3];
  const float* bk = (const float*)d_in[4];
  const float* Wv = (const float*)d_in[5];
  const float* bv = (const float*)d_in[6];
  const float* Ws = (const float*)d_in[7];
  const float* bs = (const float*)d_in[8];
  float* out = (float*)d_out;

  char* ws = (char*)d_ws;
  _Float16* Xh = (_Float16*)(ws + OFF_XH);
  _Float16* qh = (_Float16*)(ws + OFF_QH);
  _Float16* kh = (_Float16*)(ws + OFF_KH);
  _Float16* vt = (_Float16*)(ws + OFF_VT);
  _Float16* P  = (_Float16*)(ws + OFF_P);
  _Float16* Wt = (_Float16*)(ws + OFF_WT);
  float*    S  = (float*)(ws + OFF_S);

  // 1) zero padded f16 staging (Xh, qh, kh, vt, P)
  k_zero<<<2048, 256, 0, stream>>>((uint4*)ws, (long)(ZERO_BYTES / 16));
  // 2) f32 -> f16 conversions (W transposed)
  k_convert<<<2048, 256, 0, stream>>>(X, Wq, Wk, Wv, Xh, Wt);
  // 3) QKV projection GEMM (WMMA), bias + per-head scatter (V transposed)
  k_qkv<<<dim3(MPAD / 16, NHEAD, 3), 32, 0, stream>>>(Xh, Wt, bq, bk, bv,
                                                      qh, kh, vt);
  // 4) logits S = Q K^T (WMMA, A reused across 14 column tiles)
  k_logits<<<dim3(13, BATCH * NHEAD), 32, 0, stream>>>(qh, kh, S);
  // 5) 3-branch softmax combine (VALU: exp + tanh)
  k_combine<<<dim3(NTOK, BATCH * NHEAD), 256, 0, stream>>>(S, qh, kh, Ws, bs, P);
  // 6) ctx = P V (WMMA, 16x64 per wave), write f32 output
  k_ctx<<<dim3(13, BATCH * NHEAD), 32, 0, stream>>>(P, vt, out);
}
